// SNNLif_70231305225203
// MI455X (gfx1250) — matrix operand order repaired and checked by reading, not compiled, not run on hardware
//
#include <hip/hip_runtime.h>

// ---------------------------------------------------------------------------
// LIF recurrent SNN forward on gfx1250 (CDNA5, wave32, WMMA f16->f32).
// ---------------------------------------------------------------------------

typedef __attribute__((ext_vector_type(16))) _Float16 v16h;
typedef __attribute__((ext_vector_type(8)))  _Float16 v8h;
typedef __attribute__((ext_vector_type(8)))  float    v8f;

constexpr int T = 2000, B = 64, F = 128, H = 512, O = 64;
constexpr float DT_TAU_MEM = 0.1f;              // DT * TAU_MEM_INV
constexpr float DT_TAU_SYN = 0.2f;              // DT * TAU_SYN_INV
constexpr float V_TH       = 1.0f;
constexpr float K_FILT     = 0.2231435511314f;  // DT * TAU_FILTER_INV

// ---------------------------------------------------------------------------
// Kernel 0a: one-time f32 -> f16 conversion of the three weight matrices.
// ---------------------------------------------------------------------------
__global__ void convert_weights_kernel(const float* __restrict__ w_in,
                                       const float* __restrict__ w_rec,
                                       const float* __restrict__ w_out,
                                       _Float16* __restrict__ win_h,
                                       _Float16* __restrict__ wrec_h,
                                       _Float16* __restrict__ wout_h) {
  const int stride = gridDim.x * blockDim.x;
  const int n_in = H * F, n_rec = H * H, n_out = O * H;
  for (int i = blockIdx.x * blockDim.x + threadIdx.x; i < n_rec; i += stride) {
    wrec_h[i] = (_Float16)w_rec[i];
    if (i < n_in)  win_h[i]  = (_Float16)w_in[i];
    if (i < n_out) wout_h[i] = (_Float16)w_out[i];
  }
}

// ---------------------------------------------------------------------------
// Kernel 0b: one-time f32 -> f16 conversion of x (keeps the scan's critical
// path free of v_cvt chains; A fragments become straight b128 loads).
// ---------------------------------------------------------------------------
__global__ void convert_x_kernel(const float* __restrict__ x,
                                 _Float16* __restrict__ x_h) {
  const size_t n = (size_t)T * B * F;
  const size_t stride = (size_t)gridDim.x * blockDim.x;
  for (size_t i = blockIdx.x * (size_t)blockDim.x + threadIdx.x; i < n; i += stride)
    x_h[i] = (_Float16)x[i];
}

// Build a 16-half A/B fragment from two contiguous 8-half (16B) runs.
__device__ inline v16h make_frag(v8h lo, v8h hi) {
  return __builtin_shufflevector(lo, hi, 0, 1, 2, 3, 4, 5, 6, 7,
                                 8, 9, 10, 11, 12, 13, 14, 15);
}

// ---------------------------------------------------------------------------
// Kernel 1: fused input-GEMM + recurrent-GEMM + LIF scan.
// One workgroup per 16-row batch tile; 32 waves, one 16-wide H tile per wave.
// v/i state lives in C-fragment VGPRs for the whole 2000-step scan; spikes z
// are double-buffered in LDS (f16) and also streamed to global for kernel 2.
// 4 round-robin accumulators keep the WMMA dependency chain at 5, not 20.
// ---------------------------------------------------------------------------
__global__ __launch_bounds__(1024)
void lif_recurrent_kernel(const _Float16* __restrict__ x_h,     // [T][B][F]
                          const _Float16* __restrict__ win_h,   // [H][F]
                          const _Float16* __restrict__ wrec_h,  // [H][H]
                          _Float16* __restrict__ z_out,         // [T][B][H]
                          float* __restrict__ zT,
                          float* __restrict__ vT,
                          float* __restrict__ iT) {
  __shared__ _Float16 zbuf[2][16][H];   // 32 KB spike double buffer

  const int wave  = threadIdx.x >> 5;   // 0..31 (wave32)
  const int lane  = threadIdx.x & 31;
  const int b0    = blockIdx.x * 16;    // batch tile base
  const int h0    = wave * 16;          // this wave's N tile of H
  const int upper = (lane >= 16);       // lane-half selector
  const int mrow  = lane & 15;          // A-row (M) / B-col (N) index

  // z(t=0) = 0
  for (int idx = threadIdx.x; idx < 2 * 16 * H; idx += blockDim.x)
    ((_Float16*)zbuf)[idx] = (_Float16)0.0f;
  __syncthreads();

  v8f vmem = {};   // membrane potential fragment (C layout)
  v8f icur = {};   // synaptic current fragment

  // B-fragment rows: column N of B == row (h0+mrow) of the weight matrix.
  const _Float16* wrec_row0 = wrec_h + (size_t)(h0 + mrow) * H;
  const _Float16* win_row0  = win_h  + (size_t)(h0 + mrow) * F;
  const int aoff = upper ? 8 : 0;    // A fragment K sub-offset
  const int boff = upper ? 8 : 0;    // B fragment K sub-offset

  for (int t = 0; t < T; ++t) {
    const int cur = t & 1;
    const int nxt = cur ^ 1;

    // Opaque zero offset, regenerated every iteration: the B-fragment loads
    // below depend on it, so LICM cannot hoist (and then spill) the
    // loop-invariant weight fragments out of the 2000-step scan.  Unlike
    // laundering the *pointer* through asm, this keeps the address chain
    // kernel-arg-derived, so clang still emits global_load_b128 (not flat_*),
    // keeping the weight loads on LOADcnt only (decoupled from DScnt).
    int zoff = 0;
    asm volatile("" : "+v"(zoff));
    const _Float16* wr = wrec_row0 + zoff;
    const _Float16* wi = win_row0 + zoff;

    const _Float16* xrow = x_h + ((size_t)t * B + (b0 + mrow)) * F;
    v8f acc[4] = {{}, {}, {}, {}};

    // ---- GEMM 1: acc += x_t[16xF] * w_in^T  (K = F = 128, 4 WMMA) ----
#pragma unroll
    for (int kk = 0; kk < 4; ++kk) {
      const int k  = kk * 32;
      const int o1 = k + aoff;                // A: K runs {o1.., o1+16..}
      v16h a  = make_frag(*(const v8h*)(xrow + o1),
                          *(const v8h*)(xrow + o1 + 16));
      const int ks = k + boff;                // B: K runs {ks.., ks+16..}
      v16h bf = make_frag(*(const v8h*)(wi + ks),
                          *(const v8h*)(wi + ks + 16));
      acc[kk] = __builtin_amdgcn_wmma_f32_16x16x32_f16(
          false, a, false, bf, (short)0, acc[kk], false, false);
    }

    // ---- GEMM 2: acc += z_prev[16xH] * w_rec^T  (K = H = 512, 16 WMMA) ----
#pragma unroll
    for (int kk = 0; kk < 16; ++kk) {
      const int k  = kk * 32;
      const int o1 = k + aoff;
      v16h a = make_frag(*(const v8h*)(&zbuf[cur][mrow][o1]),
                         *(const v8h*)(&zbuf[cur][mrow][o1 + 16]));
      const int ks = k + boff;
      if (kk + 1 < 16)   // pull next w_rec chunk into near caches (WGP scope)
        __builtin_prefetch(wr + ks + 32, 0, 3);
      v16h bf = make_frag(*(const v8h*)(wr + ks),
                          *(const v8h*)(wr + ks + 16));
      acc[kk & 3] = __builtin_amdgcn_wmma_f32_16x16x32_f16(
          false, a, false, bf, (short)0, acc[kk & 3], false, false);
    }

    const v8f accsum = (acc[0] + acc[1]) + (acc[2] + acc[3]);

    // ---- LIF update directly on the C fragment ----
    // C layout: element r of v8f -> row M = r + (upper?8:0), col N = mrow.
#pragma unroll
    for (int r = 0; r < 8; ++r) {
      const float v_dec = vmem[r] + DT_TAU_MEM * (icur[r] - vmem[r]);
      const float i_dec = icur[r] * (1.0f - DT_TAU_SYN);
      const float z     = (v_dec - V_TH > 0.0f) ? 1.0f : 0.0f;
      vmem[r] = (1.0f - z) * v_dec;     // reset to V_RESET = 0 on spike
      icur[r] = i_dec + accsum[r];
      const _Float16 zh = (_Float16)z;  // exact: z in {0,1}
      const int m = r + aoff;           // row in tile (aoff == upper*8)
      zbuf[nxt][m][h0 + mrow] = zh;
      z_out[((size_t)t * B + (b0 + m)) * H + h0 + mrow] = zh;
      if (t == T - 1)
        zT[(size_t)(b0 + m) * H + h0 + mrow] = z;
    }
    __syncthreads();   // single barrier per step (double-buffered z)
  }

  // Final membrane/current state in f32
#pragma unroll
  for (int r = 0; r < 8; ++r) {
    const int m = r + aoff;
    const size_t idx = (size_t)(b0 + m) * H + h0 + mrow;
    vT[idx] = vmem[r];
    iT[idx] = icur[r];
  }
}

// ---------------------------------------------------------------------------
// Kernel 2: u[tb][o] = z[tb][:] @ w_out^T + b_out   (one 16x16 tile per wave)
// ---------------------------------------------------------------------------
__global__ __launch_bounds__(256)
void out_gemm_kernel(const _Float16* __restrict__ z_h,     // [T*B][H]
                     const _Float16* __restrict__ wout_h,  // [O][H]
                     const float* __restrict__ b_out,      // [O]
                     float* __restrict__ u) {              // [T*B][O]
  const int gwave = (int)((blockIdx.x * blockDim.x + threadIdx.x) >> 5);
  const int lane  = threadIdx.x & 31;
  const int n_ct  = O / 16;                      // 4 column tiles
  const int row_t = gwave / n_ct;
  const int col_t = gwave % n_ct;
  if (row_t >= (T * B) / 16) return;
  const int r0 = row_t * 16, o0 = col_t * 16;
  const int upper = (lane >= 16);
  const int mrow  = lane & 15;
  const int aoff  = upper ? 8 : 0;
  const int boff  = upper ? 8 : 0;

  const _Float16* zrow = z_h    + (size_t)(r0 + mrow) * H;
  const _Float16* wrow = wout_h + (size_t)(o0 + mrow) * H;
  v8f acc0 = {}, acc1 = {};
#pragma unroll
  for (int kk = 0; kk < 16; ++kk) {
    const int k  = kk * 32;
    const int o1 = k + aoff;
    v16h a  = make_frag(*(const v8h*)(zrow + o1),
                        *(const v8h*)(zrow + o1 + 16));
    const int ks = k + boff;
    v16h bf = make_frag(*(const v8h*)(wrow + ks),
                        *(const v8h*)(wrow + ks + 16));
    if (kk & 1)
      acc1 = __builtin_amdgcn_wmma_f32_16x16x32_f16(false, a, false, bf,
                                                    (short)0, acc1, false, false);
    else
      acc0 = __builtin_amdgcn_wmma_f32_16x16x32_f16(false, a, false, bf,
                                                    (short)0, acc0, false, false);
  }
  const v8f acc = acc0 + acc1;
  const float bo = b_out[o0 + mrow];   // column N = mrow
#pragma unroll
  for (int r = 0; r < 8; ++r) {
    const int m = r + aoff;
    u[(size_t)(r0 + m) * O + o0 + mrow] = acc[r] + bo;
  }
}

// ---------------------------------------------------------------------------
// Kernel 3: exponential filter scan over T (per (b,o) element, coalesced in t)
// ---------------------------------------------------------------------------
__global__ void exp_filter_kernel(const float* __restrict__ u,
                                  float* __restrict__ ys) {
  const int idx = blockIdx.x * blockDim.x + threadIdx.x;   // 0..B*O-1
  if (idx >= B * O) return;
  float y = u[idx];          // y[0] = u[0]
  ys[idx] = y;
  for (int t = 1; t < T; ++t) {
    const float ut = u[(size_t)t * B * O + idx];
    y = y + K_FILT * (ut - y);
    ys[(size_t)t * B * O + idx] = y;
  }
}

// ---------------------------------------------------------------------------
extern "C" void kernel_launch(void* const* d_in, const int* in_sizes, int n_in,
                              void* d_out, int out_size, void* d_ws, size_t ws_size,
                              hipStream_t stream) {
  (void)in_sizes; (void)n_in; (void)out_size; (void)ws_size;
  const float* x     = (const float*)d_in[0];   // [T,B,F]
  const float* w_in  = (const float*)d_in[1];   // [H,F]
  const float* w_rec = (const float*)d_in[2];   // [H,H]
  const float* w_out = (const float*)d_in[3];   // [O,H]
  const float* b_out = (const float*)d_in[4];   // [O]

  // Workspace carve-out (256B aligned)
  char* ws = (char*)d_ws;
  size_t off = 0;
  auto carve = [&](size_t bytes) -> char* {
    char* p = ws + off;
    off = (off + bytes + 255) & ~(size_t)255;
    return p;
  };
  _Float16* wrec_h = (_Float16*)carve((size_t)H * H * 2);
  _Float16* win_h  = (_Float16*)carve((size_t)H * F * 2);
  _Float16* wout_h = (_Float16*)carve((size_t)O * H * 2);
  _Float16* x_h    = (_Float16*)carve((size_t)T * B * F * 2);   // ~33 MB
  _Float16* z_h    = (_Float16*)carve((size_t)T * B * H * 2);   // ~131 MB
  float*    u      = (float*)   carve((size_t)T * B * O * 4);   // ~33 MB

  // Output layout: ys [T,B,O] then zT, vT, iT [B,H] each
  float* ys = (float*)d_out;
  float* zT = ys + (size_t)T * B * O;
  float* vT = zT + (size_t)B * H;
  float* iT = vT + (size_t)B * H;

  convert_weights_kernel<<<512, 256, 0, stream>>>(w_in, w_rec, w_out,
                                                  win_h, wrec_h, wout_h);
  convert_x_kernel<<<2048, 256, 0, stream>>>(x, x_h);
  lif_recurrent_kernel<<<B / 16, 1024, 0, stream>>>(x_h, win_h, wrec_h,
                                                    z_h, zT, vT, iT);
  const int waves  = ((T * B) / 16) * (O / 16);        // 32000 tiles
  const int blocks = (waves * 32 + 255) / 256;
  out_gemm_kernel<<<blocks, 256, 0, stream>>>(z_h, wout_h, b_out, u);
  exp_filter_kernel<<<(B * O + 255) / 256, 256, 0, stream>>>(u, ys);
}
